// DeformableTransformer_78211354460371
// MI455X (gfx1250) — matrix operand order repaired and checked by reading, not compiled, not run on hardware
//
#include <hip/hip_runtime.h>
#include <hip/hip_bf16.h>

// ---------------------------------------------------------------------------
// Deformable-DETR transformer forward for gfx1250 (MI455X).
// GEMMs use v_wmma_f32_16x16x32_bf16 with pre-swizzled bf16 weight fragments
// (coalesced b128 loads); sampler/LN/MHA are wave32-shaped.
// ---------------------------------------------------------------------------

typedef __attribute__((ext_vector_type(16))) __bf16        bf16x16;
typedef __attribute__((ext_vector_type(8)))  float         f32x8;
typedef __attribute__((ext_vector_type(8)))  unsigned int  u32x8;

#define CDM   256      // d_model
#define NHEAD 8
#define NQ    300
#define BATCH 2
#define S_TOT 8500
#define ME      17024  // padded encoder rows (2*8500 -> 17024 = 532*32)
#define ME_REAL 17000
#define MD      608    // padded decoder rows (2*300 -> 608 = 19*32)
#define MD_REAL 600

__device__ __forceinline__ float wave_sum(float v) {
#pragma unroll
  for (int m = 16; m >= 1; m >>= 1) v += __shfl_xor(v, m, 32);
  return v;
}
__device__ __forceinline__ float wave_max(float v) {
#pragma unroll
  for (int m = 16; m >= 1; m >>= 1) v = fmaxf(v, __shfl_xor(v, m, 32));
  return v;
}

// ---------------------------------------------------------------------------
// Weight pre-swizzle: W[K,N] f32 (ldw) -> packed bf16 B-fragments.
// Fragment f = kb*(N/16) + nt covers K-block kb (32) x N-tile nt (16).
// Lane L element j = W[kb*32 + (L/16)*16 + j][nt*16 + L%16], j=0..15,
// packed 2 bf16 per dword -> out[((f*32 + L)*8 + d)].
// ---------------------------------------------------------------------------
__global__ void conv_w_kernel(const float* __restrict__ W, int ldw, int K, int N,
                              unsigned* __restrict__ out) {
  int idx = blockIdx.x * blockDim.x + threadIdx.x;
  int total = (K * N) >> 1;
  if (idx >= total) return;
  int d  = idx & 7;
  int L  = (idx >> 3) & 31;
  int f  = idx >> 8;
  int nT = N >> 4;
  int kb = f / nT, nt = f - kb * nT;
  int half = L >> 4, l16 = L & 15;
  int k = kb * 32 + half * 16 + (d << 1);
  int n = (nt << 4) + l16;
  float lo = W[(size_t)k * ldw + n];
  float hi = W[(size_t)(k + 1) * ldw + n];
  __bf16 a = (__bf16)lo, b = (__bf16)hi;
  unsigned short ua = __builtin_bit_cast(unsigned short, a);
  unsigned short ub = __builtin_bit_cast(unsigned short, b);
  out[idx] = ((unsigned)ub << 16) | (unsigned)ua;
}

// ---------------------------------------------------------------------------
// GEMM: C[M,N] = A[M,K](f32,lda) * W[K,N](pre-swizzled bf16 frags) + bias.
// Each wave: 32(M) x 64(N) tile = 8 accumulators; B frags reused across the
// two M-tiles. M%32==0, N%64==0, K%32==0.
// ---------------------------------------------------------------------------
__global__ void gemm_wmma_kernel(const float* __restrict__ A, int lda,
                                 const unsigned* __restrict__ Wf,
                                 const float* __restrict__ bias,
                                 float* __restrict__ C, int ldc,
                                 int M, int N, int K, int relu) {
  const int lane = threadIdx.x & 31;
  const int wid  = threadIdx.x >> 5;
  const int nStrips = N >> 6;
  const int mBlocks = M >> 5;
  int tile = blockIdx.x * (blockDim.x >> 5) + wid;
  if (tile >= mBlocks * nStrips) return;
  const int mb = tile / nStrips;
  const int ns = tile - mb * nStrips;
  const int m0 = mb << 5;
  const int n0 = ns << 6;
  const int half = lane >> 4;   // lane 0-15 vs 16-31
  const int l16  = lane & 15;
  const int nT16 = N >> 4;
  const int nt0  = n0 >> 4;

  f32x8 acc[2][4];
#pragma unroll
  for (int mi = 0; mi < 2; ++mi)
#pragma unroll
    for (int t = 0; t < 4; ++t)
#pragma unroll
      for (int r = 0; r < 8; ++r) acc[mi][t][r] = 0.0f;

  const float* arow0 = A + (size_t)(m0 + l16) * lda + half * 16;
  const float* arow1 = arow0 + (size_t)16 * lda;
  const int kBlocks = K >> 5;

  for (int kb = 0; kb < kBlocks; ++kb) {
    // A fragments: 16 consecutive-f32 chunks -> 4x float4 each
    bf16x16 af[2];
#pragma unroll
    for (int mi = 0; mi < 2; ++mi) {
      const float* ar = (mi ? arow1 : arow0) + (kb << 5);
      float4 p0 = *(const float4*)(ar);
      float4 p1 = *(const float4*)(ar + 4);
      float4 p2 = *(const float4*)(ar + 8);
      float4 p3 = *(const float4*)(ar + 12);
      af[mi][0]  = (__bf16)p0.x; af[mi][1]  = (__bf16)p0.y;
      af[mi][2]  = (__bf16)p0.z; af[mi][3]  = (__bf16)p0.w;
      af[mi][4]  = (__bf16)p1.x; af[mi][5]  = (__bf16)p1.y;
      af[mi][6]  = (__bf16)p1.z; af[mi][7]  = (__bf16)p1.w;
      af[mi][8]  = (__bf16)p2.x; af[mi][9]  = (__bf16)p2.y;
      af[mi][10] = (__bf16)p2.z; af[mi][11] = (__bf16)p2.w;
      af[mi][12] = (__bf16)p3.x; af[mi][13] = (__bf16)p3.y;
      af[mi][14] = (__bf16)p3.z; af[mi][15] = (__bf16)p3.w;
    }
#pragma unroll
    for (int t = 0; t < 4; ++t) {
      const u32x8* bp = (const u32x8*)(Wf +
          (((size_t)(kb * nT16 + nt0 + t) * 32 + lane) << 3));
      bf16x16 bf = __builtin_bit_cast(bf16x16, *bp);
      acc[0][t] = __builtin_amdgcn_wmma_f32_16x16x32_bf16(
          false, af[0], false, bf, (short)0, acc[0][t], false, false);
      acc[1][t] = __builtin_amdgcn_wmma_f32_16x16x32_bf16(
          false, af[1], false, bf, (short)0, acc[1][t], false, false);
    }
  }

#pragma unroll
  for (int mi = 0; mi < 2; ++mi)
#pragma unroll
    for (int t = 0; t < 4; ++t) {
      const int n = n0 + t * 16 + l16;
      const float bv = bias ? bias[n] : 0.0f;
#pragma unroll
      for (int r = 0; r < 8; ++r) {
        float v = acc[mi][t][r] + bv;
        if (relu) v = fmaxf(v, 0.0f);
        C[(size_t)(m0 + mi * 16 + half * 8 + r) * ldc + n] = v;
      }
    }
}

// ---------------------------------------------------------------------------
// Elementwise add: out = a + b
// ---------------------------------------------------------------------------
__global__ void add_kernel(float* __restrict__ out, const float* __restrict__ a,
                           const float* __restrict__ b, int n) {
  int i = blockIdx.x * blockDim.x + threadIdx.x;
  if (i < n) out[i] = a[i] + b[i];
}

// ---------------------------------------------------------------------------
// Encoder prep: x = padded src, pos = pos_embed + level_embed, grid ref points
// ---------------------------------------------------------------------------
__global__ void enc_prep_kernel(const float* __restrict__ src,
                                const float* __restrict__ pos_embed,
                                const float* __restrict__ level_embed,
                                const float* __restrict__ vr,
                                float* __restrict__ x, float* __restrict__ pos,
                                float* __restrict__ grid) {
  int idx = blockIdx.x * blockDim.x + threadIdx.x;
  if (idx >= ME * CDM) return;
  int t = idx >> 8;
  int c = idx & 255;
  if (t < ME_REAL) {
    int s = t % S_TOT;
    int lvl = (s < 6400) ? 0 : (s < 8000) ? 1 : (s < 8400) ? 2 : 3;
    x[idx]   = src[(size_t)t * CDM + c];
    pos[idx] = pos_embed[(size_t)t * CDM + c] + level_embed[lvl * CDM + c];
    if (c < 2) {
      const int LW[4] = {80, 40, 20, 10};
      const int LS[4] = {0, 6400, 8000, 8400};
      int b   = t / S_TOT;
      int loc = s - LS[lvl];
      int iy  = loc / LW[lvl];
      int ix  = loc - iy * LW[lvl];
      float g;
      if (c == 0) g = (ix + 0.5f) / (vr[(b * 4 + lvl) * 2 + 0] * LW[lvl]);
      else        g = (iy + 0.5f) / (vr[(b * 4 + lvl) * 2 + 1] * LW[lvl]);
      grid[t * 2 + c] = g;
    }
  } else {
    x[idx] = 0.0f;
    pos[idx] = 0.0f;
  }
}

// ---------------------------------------------------------------------------
// Softmax over groups of 16 (per head, L*P=16): in-place on [Mreal, 128]
// ---------------------------------------------------------------------------
__global__ void softmax16_kernel(float* __restrict__ a, int Mreal) {
  int idx = blockIdx.x * blockDim.x + threadIdx.x;
  if (idx >= Mreal * NHEAD) return;
  float* p = a + (size_t)idx * 16;
  float mx = -1e30f;
#pragma unroll
  for (int j = 0; j < 16; ++j) mx = fmaxf(mx, p[j]);
  float s = 0.0f;
  float e[16];
#pragma unroll
  for (int j = 0; j < 16; ++j) { e[j] = __expf(p[j] - mx); s += e[j]; }
  float inv = 1.0f / s;
#pragma unroll
  for (int j = 0; j < 16; ++j) p[j] = e[j] * inv;
}

// ---------------------------------------------------------------------------
// MSDA core: one wave per (token, head), lane = channel (Dh == 32 == wave32)
// ---------------------------------------------------------------------------
__global__ void msda_sample_kernel(const float* __restrict__ value,
                                   const float* __restrict__ attn,
                                   const float* __restrict__ off,
                                   const float* __restrict__ refbase,
                                   const float* __restrict__ vr,
                                   float* __restrict__ out,
                                   int Lq, int Mreal) {
  int w = blockIdx.x * (blockDim.x >> 5) + (threadIdx.x >> 5);
  if (w >= Mreal * NHEAD) return;
  int t    = w >> 3;
  int h    = w & 7;
  int lane = threadIdx.x & 31;
  int b    = t / Lq;
  const int LW[4] = {80, 40, 20, 10};
  const int LS[4] = {0, 6400, 8000, 8400};
  float rx = refbase[t * 2 + 0];
  float ry = refbase[t * 2 + 1];
  float acc = 0.0f;
#pragma unroll
  for (int l = 0; l < 4; ++l) {
    const int W_ = LW[l], H_ = LW[l];
    const float lx = rx * vr[(b * 4 + l) * 2 + 0] * W_;
    const float ly = ry * vr[(b * 4 + l) * 2 + 1] * H_;
    const float* vbase = value + (size_t)(b * S_TOT + LS[l]) * CDM + h * 32 + lane;
    const float* ob = off  + (size_t)t * 256 + h * 32 + l * 8;
    const float* ab = attn + (size_t)t * 128 + h * 16 + l * 4;
#pragma unroll
    for (int p = 0; p < 4; ++p) {
      float x = lx + ob[p * 2 + 0] - 0.5f;
      float y = ly + ob[p * 2 + 1] - 0.5f;
      float a = ab[p];
      float xf = floorf(x), yf = floorf(y);
      int x0 = (int)xf, y0 = (int)yf;
      float wx = x - xf, wy = y - yf;
      float s = 0.0f;
#pragma unroll
      for (int cy = 0; cy < 2; ++cy)
#pragma unroll
        for (int cx = 0; cx < 2; ++cx) {
          int iy = y0 + cy, ix = x0 + cx;
          bool valid = (iy >= 0) && (iy < H_) && (ix >= 0) && (ix < W_);
          int iyc = min(max(iy, 0), H_ - 1);
          int ixc = min(max(ix, 0), W_ - 1);
          float v = vbase[(size_t)(iyc * W_ + ixc) * CDM];
          float wgt = (cx ? wx : 1.0f - wx) * (cy ? wy : 1.0f - wy);
          s += valid ? v * wgt : 0.0f;
        }
      acc += a * s;
    }
  }
  out[(size_t)t * CDM + h * 32 + lane] = acc;
}

// ---------------------------------------------------------------------------
// LayerNorm with residual: out = LN(x + res) * g + b, one wave per row (C=256)
// ---------------------------------------------------------------------------
__global__ void ln_kernel(const float* __restrict__ x, const float* __restrict__ res,
                          const float* __restrict__ g, const float* __restrict__ bta,
                          float* __restrict__ out, int Mreal) {
  int row = blockIdx.x * (blockDim.x >> 5) + (threadIdx.x >> 5);
  if (row >= Mreal) return;
  int lane = threadIdx.x & 31;
  const float* xr = x + (size_t)row * CDM;
  const float* rr = res ? res + (size_t)row * CDM : nullptr;
  float v[8];
  float s = 0.0f;
#pragma unroll
  for (int i = 0; i < 8; ++i) {
    float t = xr[lane + i * 32];
    if (rr) t += rr[lane + i * 32];
    v[i] = t;
    s += t;
  }
  s = wave_sum(s);
  float mean = s * (1.0f / CDM);
  float q = 0.0f;
#pragma unroll
  for (int i = 0; i < 8; ++i) { float d = v[i] - mean; q += d * d; }
  q = wave_sum(q);
  float inv = rsqrtf(q * (1.0f / CDM) + 1e-5f);
  float* orow = out + (size_t)row * CDM;
#pragma unroll
  for (int i = 0; i < 8; ++i) {
    int c = lane + i * 32;
    orow[c] = (v[i] - mean) * inv * g[c] + bta[c];
  }
}

// ---------------------------------------------------------------------------
// Decoder MHA: one wave (block of 32) per (b, h, q). qk: [MD,512] (qp|kp),
// v: [MD,256]. probs staged in LDS.
// ---------------------------------------------------------------------------
__global__ void mha_kernel(const float* __restrict__ qk, const float* __restrict__ v,
                           float* __restrict__ out) {
  int idx = blockIdx.x;              // b*2400 + h*300 + q
  int q = idx % NQ;
  int h = (idx / NQ) % NHEAD;
  int b = idx / (NQ * NHEAD);
  int t = b * NQ + q;
  __shared__ float qvec[32];
  __shared__ float probs[NQ];
  int lane = threadIdx.x;
  qvec[lane] = qk[(size_t)t * 512 + h * 32 + lane];
  __syncthreads();
  const float scale = 0.17677669529663687f;  // 1/sqrt(32)
  float sc[10];
  float lmax = -1e30f;
  int nk = 0;
  for (int k = lane; k < NQ; k += 32) {
    const float* kr = qk + (size_t)(b * NQ + k) * 512 + 256 + h * 32;
    float s = 0.0f;
#pragma unroll
    for (int d = 0; d < 32; ++d) s += qvec[d] * kr[d];
    s *= scale;
    sc[nk++] = s;
    lmax = fmaxf(lmax, s);
  }
  lmax = wave_max(lmax);
  float lsum = 0.0f;
  nk = 0;
  for (int k = lane; k < NQ; k += 32) {
    float e = __expf(sc[nk++] - lmax);
    probs[k] = e;
    lsum += e;
  }
  lsum = wave_sum(lsum);
  __syncthreads();
  float inv = 1.0f / lsum;
  float acc = 0.0f;
  for (int k = 0; k < NQ; ++k)
    acc += probs[k] * v[(size_t)(b * NQ + k) * CDM + h * 32 + lane];
  out[(size_t)t * CDM + h * 32 + lane] = acc * inv;
}

// ---------------------------------------------------------------------------
// Decoder prep: split query_embed -> qpos (first 256) and tgt (last 256)
// ---------------------------------------------------------------------------
__global__ void dec_prep_kernel(const float* __restrict__ query_embed,
                                float* __restrict__ qpos, float* __restrict__ x) {
  int idx = blockIdx.x * blockDim.x + threadIdx.x;
  if (idx >= MD * CDM) return;
  int t = idx >> 8;
  int c = idx & 255;
  if (t < MD_REAL) {
    int q = t % NQ;
    qpos[idx] = query_embed[(size_t)q * 512 + c];
    x[idx]    = query_embed[(size_t)q * 512 + 256 + c];
  } else {
    qpos[idx] = 0.0f;
    x[idx] = 0.0f;
  }
}

// ref = sigmoid(qpos @ ref_W + ref_b)  : one thread per (t, c) of [600, 2]
__global__ void dec_ref_kernel(const float* __restrict__ query_embed,
                               const float* __restrict__ ref_W,
                               const float* __restrict__ ref_b,
                               float* __restrict__ ref) {
  int idx = blockIdx.x * blockDim.x + threadIdx.x;
  if (idx >= MD_REAL * 2) return;
  int t = idx >> 1;
  int c = idx & 1;
  int q = t % NQ;
  float s = ref_b[c];
  for (int k = 0; k < CDM; ++k)
    s += query_embed[(size_t)q * 512 + k] * ref_W[k * 2 + c];
  ref[idx] = 1.0f / (1.0f + expf(-s));
}

// d_out = concat(out flat [2*300*256], ref flat [2*300*2])
__global__ void copyout_kernel(const float* __restrict__ x,
                               const float* __restrict__ ref,
                               float* __restrict__ out) {
  int idx = blockIdx.x * blockDim.x + threadIdx.x;
  const int n0 = MD_REAL * CDM;      // 153600
  if (idx < n0) out[idx] = x[idx];
  else if (idx < n0 + MD_REAL * 2) out[idx] = ref[idx - n0];
}

// ---------------------------------------------------------------------------
// Host-side launcher
// ---------------------------------------------------------------------------
static inline void launch_gemm(hipStream_t st, const float* A, int lda,
                               const unsigned* Wf, const float* bias,
                               float* C, int ldc, int M, int N, int K, int relu) {
  int waves  = (M / 32) * (N / 64);
  int blocks = (waves + 7) / 8;
  gemm_wmma_kernel<<<blocks, 256, 0, st>>>(A, lda, Wf, bias, C, ldc, M, N, K, relu);
}

extern "C" void kernel_launch(void* const* d_in, const int* in_sizes, int n_in,
                              void* d_out, int out_size, void* d_ws, size_t ws_size,
                              hipStream_t stream) {
  (void)in_sizes; (void)n_in; (void)out_size; (void)ws_size;
#define IN(i) ((const float*)d_in[i])
  const float* src         = IN(0);
  const float* pos_embed   = IN(1);
  const float* query_embed = IN(2);
  const float* vr          = IN(3);   // valid_ratios [2,4,2]
  const float* level_embed = IN(4);
  const float* ref_W       = IN(5);
  const float* ref_b       = IN(6);
  // enc_params (insertion order): Wv bv Woff boff Wattn battn Wout bout W1 b1 W2 b2 n1g n1b n2g n2b
  const float* eWv = IN(7),  *eBv = IN(8),  *eWoff = IN(9),  *eBoff = IN(10);
  const float* eWat = IN(11), *eBat = IN(12), *eWout = IN(13), *eBout = IN(14);
  const float* eW1 = IN(15), *eB1 = IN(16), *eW2 = IN(17), *eB2 = IN(18);
  const float* eN1g = IN(19), *eN1b = IN(20), *eN2g = IN(21), *eN2b = IN(22);
  // dec_params: same 16 + Wqkv bqkv Wo bo n3g n3b
  const float* dWv = IN(23), *dBv = IN(24), *dWoff = IN(25), *dBoff = IN(26);
  const float* dWat = IN(27), *dBat = IN(28), *dWout = IN(29), *dBout = IN(30);
  const float* dW1 = IN(31), *dB1 = IN(32), *dW2 = IN(33), *dB2 = IN(34);
  const float* dN1g = IN(35), *dN1b = IN(36), *dN2g = IN(37), *dN2b = IN(38);
  const float* dWqkv = IN(39), *dBqkv = IN(40), *dWo = IN(41), *dBo = IN(42);
  const float* dN3g = IN(43), *dN3b = IN(44);
#undef IN

  // ---- workspace carve-up (floats) ----
  float* ws = (float*)d_ws;
  size_t o = 0;
  auto alloc = [&](size_t n) { float* p = ws + o; o += n; return p; };
  float* b_pos   = alloc((size_t)ME * 256);
  float* b_x     = alloc((size_t)ME * 256);   // encoder activations -> memory
  float* b_qin   = alloc((size_t)ME * 256);   // qin, later reused as sampled
  float* b_val   = alloc((size_t)ME * 256);   // value, also GEMM temp
  float* b_off   = alloc((size_t)ME * 256);
  float* b_attn  = alloc((size_t)ME * 128);
  float* b_hid   = alloc((size_t)ME * 1024);  // FFN hidden / MHA out temp
  float* b_grid  = alloc((size_t)ME * 2);
  float* b_dqpos = alloc((size_t)MD * 256);
  float* b_dx    = alloc((size_t)MD * 256);
  float* b_dqk   = alloc((size_t)MD * 512);
  float* b_dv    = alloc((size_t)MD * 256);
  float* b_dref  = alloc((size_t)MD * 2);

  // bf16 swizzled-weight region (dwords), after the float buffers
  unsigned* wseg = (unsigned*)(ws + o);
  size_t wo = 0;
  auto convW = [&](const float* W, int ldw, int K, int N) -> const unsigned* {
    unsigned* dst = wseg + wo;
    size_t ndw = ((size_t)K * N) >> 1;
    wo += ndw;
    conv_w_kernel<<<(int)((ndw + 255) / 256), 256, 0, stream>>>(W, ldw, K, N, dst);
    return dst;
  };

  const int TPB = 256;
  auto nblk = [](int n) { return (n + 255) / 256; };

  // ---- encoder prep ----
  enc_prep_kernel<<<nblk(ME * 256), TPB, 0, stream>>>(src, pos_embed, level_embed,
                                                      vr, b_x, b_pos, b_grid);

  // ---- encoder layers ----
  for (int i = 0; i < 6; ++i) {
    const unsigned* Wvf  = convW(eWv  + (size_t)i * 256 * 256,  256, 256, 256);
    const unsigned* Woff = convW(eWoff + (size_t)i * 256 * 256, 256, 256, 256);
    const unsigned* Waf  = convW(eWat + (size_t)i * 256 * 128,  128, 256, 128);
    const unsigned* Wof2 = convW(eWout + (size_t)i * 256 * 256, 256, 256, 256);
    const unsigned* W1f  = convW(eW1 + (size_t)i * 256 * 1024, 1024, 256, 1024);
    const unsigned* W2f  = convW(eW2 + (size_t)i * 1024 * 256,  256, 1024, 256);
    const float* bv = eBv + (size_t)i * 256;
    const float* bof = eBoff + (size_t)i * 256;
    const float* ba = eBat + (size_t)i * 128;
    const float* bo = eBout + (size_t)i * 256;
    const float* b1 = eB1 + (size_t)i * 1024;
    const float* b2 = eB2 + (size_t)i * 256;

    add_kernel<<<nblk(ME_REAL * 256), TPB, 0, stream>>>(b_qin, b_x, b_pos, ME_REAL * 256);
    launch_gemm(stream, b_x,   256, Wvf,  bv,  b_val,  256, ME, 256,  256, 0);
    launch_gemm(stream, b_qin, 256, Woff, bof, b_off,  256, ME, 256,  256, 0);
    launch_gemm(stream, b_qin, 256, Waf,  ba,  b_attn, 128, ME, 128,  256, 0);
    softmax16_kernel<<<nblk(ME_REAL * 8), TPB, 0, stream>>>(b_attn, ME_REAL);
    msda_sample_kernel<<<ME_REAL, TPB, 0, stream>>>(b_val, b_attn, b_off, b_grid,
                                                    vr, b_qin, S_TOT, ME_REAL);
    launch_gemm(stream, b_qin, 256, Wof2, bo, b_val, 256, ME, 256, 256, 0);
    ln_kernel<<<(ME_REAL + 7) / 8, TPB, 0, stream>>>(b_x, b_val,
        eN1g + (size_t)i * 256, eN1b + (size_t)i * 256, b_x, ME_REAL);
    launch_gemm(stream, b_x,   256,  W1f, b1, b_hid, 1024, ME, 1024, 256, 1);
    launch_gemm(stream, b_hid, 1024, W2f, b2, b_val, 256,  ME, 256, 1024, 0);
    ln_kernel<<<(ME_REAL + 7) / 8, TPB, 0, stream>>>(b_x, b_val,
        eN2g + (size_t)i * 256, eN2b + (size_t)i * 256, b_x, ME_REAL);
  }
  // b_x now holds encoder memory [ME, 256]

  // ---- decoder prep ----
  dec_prep_kernel<<<nblk(MD * 256), TPB, 0, stream>>>(query_embed, b_dqpos, b_dx);
  dec_ref_kernel<<<nblk(MD_REAL * 2), TPB, 0, stream>>>(query_embed, ref_W, ref_b, b_dref);

  // ---- decoder layers ----
  for (int i = 0; i < 6; ++i) {
    const unsigned* Wqkf = convW(dWqkv + (size_t)i * 256 * 768,       768, 256, 512);
    const unsigned* Wvpf = convW(dWqkv + (size_t)i * 256 * 768 + 512, 768, 256, 256);
    const unsigned* Wofo = convW(dWo + (size_t)i * 256 * 256,  256, 256, 256);
    const unsigned* Wvf  = convW(dWv + (size_t)i * 256 * 256,  256, 256, 256);
    const unsigned* Woff = convW(dWoff + (size_t)i * 256 * 256, 256, 256, 256);
    const unsigned* Waf  = convW(dWat + (size_t)i * 256 * 128, 128, 256, 128);
    const unsigned* Wouf = convW(dWout + (size_t)i * 256 * 256, 256, 256, 256);
    const unsigned* W1f  = convW(dW1 + (size_t)i * 256 * 1024, 1024, 256, 1024);
    const unsigned* W2f  = convW(dW2 + (size_t)i * 1024 * 256, 256, 1024, 256);
    const float* bqkv = dBqkv + (size_t)i * 768;
    const float* bo = dBo + (size_t)i * 256;
    const float* bv = dBv + (size_t)i * 256;
    const float* bof = dBoff + (size_t)i * 256;
    const float* ba = dBat + (size_t)i * 128;
    const float* bou = dBout + (size_t)i * 256;
    const float* b1 = dB1 + (size_t)i * 1024;
    const float* b2 = dB2 + (size_t)i * 256;

    // self-attention (MHA):  q = x + qpos ; qp,kp from q ; vp from x
    add_kernel<<<nblk(MD_REAL * 256), TPB, 0, stream>>>(b_qin, b_dx, b_dqpos, MD_REAL * 256);
    launch_gemm(stream, b_qin, 256, Wqkf, bqkv,       b_dqk, 512, MD, 512, 256, 0);
    launch_gemm(stream, b_dx,  256, Wvpf, bqkv + 512, b_dv,  256, MD, 256, 256, 0);
    mha_kernel<<<BATCH * NHEAD * NQ, 32, 0, stream>>>(b_dqk, b_dv, b_hid);
    launch_gemm(stream, b_hid, 256, Wofo, bo, b_val, 256, MD, 256, 256, 0);
    ln_kernel<<<(MD_REAL + 7) / 8, TPB, 0, stream>>>(b_dx, b_val,
        dN2g + (size_t)i * 256, dN2b + (size_t)i * 256, b_dx, MD_REAL);

    // cross deformable attention over memory
    add_kernel<<<nblk(MD_REAL * 256), TPB, 0, stream>>>(b_qin, b_dx, b_dqpos, MD_REAL * 256);
    launch_gemm(stream, b_x,   256, Wvf,  bv,  b_val,  256, ME, 256, 256, 0); // value
    launch_gemm(stream, b_qin, 256, Woff, bof, b_off,  256, MD, 256, 256, 0);
    launch_gemm(stream, b_qin, 256, Waf,  ba,  b_attn, 128, MD, 128, 256, 0);
    softmax16_kernel<<<nblk(MD_REAL * 8), TPB, 0, stream>>>(b_attn, MD_REAL);
    msda_sample_kernel<<<MD_REAL, TPB, 0, stream>>>(b_val, b_attn, b_off, b_dref,
                                                    vr, b_qin, NQ, MD_REAL);
    launch_gemm(stream, b_qin, 256, Wouf, bou, b_val, 256, MD, 256, 256, 0);
    ln_kernel<<<(MD_REAL + 7) / 8, TPB, 0, stream>>>(b_dx, b_val,
        dN1g + (size_t)i * 256, dN1b + (size_t)i * 256, b_dx, MD_REAL);

    // FFN
    launch_gemm(stream, b_dx,  256,  W1f, b1, b_hid, 1024, MD, 1024, 256, 1);
    launch_gemm(stream, b_hid, 1024, W2f, b2, b_val, 256,  MD, 256, 1024, 0);
    ln_kernel<<<(MD_REAL + 7) / 8, TPB, 0, stream>>>(b_dx, b_val,
        dN3g + (size_t)i * 256, dN3b + (size_t)i * 256, b_dx, MD_REAL);
  }

  copyout_kernel<<<nblk(MD_REAL * CDM + MD_REAL * 2), TPB, 0, stream>>>(
      b_dx, b_dref, (float*)d_out);
}
